// SemanticCostCrossAttLite_14757507629441
// MI455X (gfx1250) — matrix operand order repaired and checked by reading, not compiled, not run on hardware
//
#include <hip/hip_runtime.h>
#include <hip/hip_bf16.h>

typedef __attribute__((ext_vector_type(16))) _Float16 v16h;
typedef __attribute__((ext_vector_type(8)))  _Float16 v8h;
typedef __attribute__((ext_vector_type(8)))  float    v8f;

#define SHUF16(a, b) __builtin_shufflevector((a), (b), 0,1,2,3,4,5,6,7,8,9,10,11,12,13,14,15)

#define B_   4
#define CCV  32
#define DD   48
#define HH   64
#define WW   128
#define CF   256
#define NS   128          // 8*16 source tokens
#define DQK  12
#define DV   12
#define HQ   32
#define WQ   64
#define NQ   (DD*HQ*WQ)   // 98304 queries per batch
#define PER_BATCH (CCV*DD*HH*WW)  // 12,582,912
#define EPSF 1e-5f

// ---- workspace layout (in floats) ----
#define WS_STATS_IN   0                    // 2*B  (sum, sumsq of cost)
#define WS_STATS_OUT  8                    // 2*B  (sum, sumsq of y)
#define WS_K          16                   // B*DQK*NS
#define WS_V          (16 + B_*DQK*NS)     // B*DQK*NS
#define WS_POUT       (16 + 2*B_*DQK*NS)   // B*NQ*CCV, layout [b][n][c]

// ------------------------------------------------------------------
__global__ void k_init(float* ws) {
    if (threadIdx.x < 16) ws[threadIdx.x] = 0.0f;
}

// per-batch sum/sumsq of cost (for GroupNorm-in)
__global__ __launch_bounds__(256) void k_reduce_in(const float* __restrict__ cost,
                                                   float* __restrict__ stats) {
    int b = blockIdx.y;
    const float* p = cost + (size_t)b * PER_BATCH;
    float s = 0.f, s2 = 0.f;
    for (size_t i = (size_t)blockIdx.x * blockDim.x + threadIdx.x; i < PER_BATCH;
         i += (size_t)gridDim.x * blockDim.x) {
        float x = p[i]; s += x; s2 += x * x;
    }
    __shared__ float sh[256], sh2[256];
    sh[threadIdx.x] = s; sh2[threadIdx.x] = s2; __syncthreads();
    for (int o = 128; o > 0; o >>= 1) {
        if (threadIdx.x < (unsigned)o) { sh[threadIdx.x] += sh[threadIdx.x + o];
                                         sh2[threadIdx.x] += sh2[threadIdx.x + o]; }
        __syncthreads();
    }
    if (threadIdx.x == 0) {
        atomicAdd(&stats[2 * b],     sh[0]);
        atomicAdd(&stats[2 * b + 1], sh2[0]);
    }
}

// k = w_k @ feat, v = w_v @ feat   (tiny: 2*4*12*128 outputs, dot over 256)
__global__ __launch_bounds__(256) void k_kv(const float* __restrict__ feat,
                                            const float* __restrict__ wk,
                                            const float* __restrict__ wv,
                                            float* __restrict__ kbuf,
                                            float* __restrict__ vbuf) {
    int idx = blockIdx.x * blockDim.x + threadIdx.x;
    if (idx >= 2 * B_ * DQK * NS) return;
    int which = idx / (B_ * DQK * NS);
    int r = idx % (B_ * DQK * NS);
    int b = r / (DQK * NS);
    int e = (r / NS) % DQK;
    int s = r % NS;
    const float* w = which ? wv : wk;
    const float* f = feat + (size_t)b * CF * NS + s;
    float acc = 0.f;
    for (int c = 0; c < CF; ++c) acc += w[e * CF + c] * f[(size_t)c * NS];
    (which ? vbuf : kbuf)[r] = acc;
}

// ------------------------------------------------------------------
// Fused attention: 1 wave handles 16 query rows. All WMMA operand panels are
// pre-staged zero-padded f16 in LDS so every fragment is 2x ds_load_b128.
// ------------------------------------------------------------------
__global__ __launch_bounds__(128) void k_attn(
        const float* __restrict__ cost, const float* __restrict__ wq,
        const float* __restrict__ wout, const float* __restrict__ gnw,
        const float* __restrict__ gnb, const float* __restrict__ stats,
        const float* __restrict__ kbuf, const float* __restrict__ vbuf,
        float* __restrict__ pout) {
    __shared__ v8h  k_h8[512];        // k_h[s][e]: 128 x 32 halves, e>=12 zero
    __shared__ v8h  v_h8[256];        // v_h[d][s]: 16 x 128 halves, d>=12 zero
    __shared__ v8h  q_h8[256];        // q_h[wave][n][e]: 4 x 16 x 32 halves, e>=12 zero
    __shared__ v8h  wo_h8[128];       // wo_h[c][d]: 32 x 32 halves, d>=12 zero
    __shared__ float xq_sh[4][16 * CCV];

    _Float16* k_hp  = (_Float16*)k_h8;
    _Float16* v_hp  = (_Float16*)v_h8;
    _Float16* q_hp  = (_Float16*)q_h8;
    _Float16* wo_hp = (_Float16*)wo_h8;

    const int lane = threadIdx.x & 31;
    const int wave = threadIdx.x >> 5;
    const int tile = blockIdx.x * 4 + wave;      // 6144 tiles/batch, 4 | 6144 -> same b per block
    const int b    = tile / (NQ / 16);
    const int n0   = (tile % (NQ / 16)) * 16;
    const int nloc = lane & 15;
    const int hi   = (lane >= 16) ? 1 : 0;       // half-lane select

    // ---- stage zero-padded f16 panels (block-wide, once) ----
    for (int i = threadIdx.x; i < 128 * 32; i += 128) {        // k_h[s][e]
        int s = i >> 5, e = i & 31;
        k_hp[i] = (_Float16)((e < DQK) ? kbuf[b * DQK * NS + e * NS + s] : 0.f);
    }
    for (int i = threadIdx.x; i < 16 * 128; i += 128) {        // v_h[d][s]
        int d = i >> 7, s = i & 127;
        v_hp[i] = (_Float16)((d < DV) ? vbuf[b * DQK * NS + d * NS + s] : 0.f);
    }
    for (int i = threadIdx.x; i < 32 * 32; i += 128) {         // wo_h[c][d]
        int c = i >> 5, d = i & 31;
        wo_hp[i] = (_Float16)((d < DV) ? wout[c * DV + d] : 0.f);
    }

    // ---- GroupNorm-in stats ----
    const float cnt  = (float)PER_BATCH;
    const float mean = stats[2 * b] / cnt;
    const float var  = stats[2 * b + 1] / cnt - mean * mean;
    const float rs   = rsqrtf(var + EPSF);

    // ---- stage pooled+normalized xq tile: 16 rows x 32 channels (per wave) ----
    for (int i = lane; i < 16 * CCV; i += 32) {
        int row = i >> 5, c = i & 31;
        int n = n0 + row;
        int d = n / (HQ * WQ); int rem = n % (HQ * WQ);
        int hq = rem / WQ;     int wqi = rem % WQ;
        const float* cp = cost + ((((size_t)b * CCV + c) * DD + d) * HH + 2 * hq) * WW + 2 * wqi;
        float a = 0.25f * (cp[0] + cp[1] + cp[WW] + cp[WW + 1]);
        xq_sh[wave][i] = (a - mean) * rs * gnw[c] + gnb[c];
    }
    __syncthreads();

    // ---- q panel: all 32 lanes share the 192 valid dot products; pad is pre-zeroed ----
    for (int i = lane; i < 16 * 32; i += 32)
        q_hp[wave * 512 + i] = (_Float16)0.f;
    const float scale = 0.28867513459481287f;    // 12^-0.5
    for (int i = lane; i < 16 * DQK; i += 32) {
        int n = i / DQK, e = i % DQK;
        const float* xr = &xq_sh[wave][n * CCV];
        float qv = 0.f;
        for (int c = 0; c < CCV; ++c) qv += wq[e * CCV + c] * xr[c];
        q_hp[wave * 512 + n * 32 + e] = (_Float16)(scale * qv);
    }
    __syncthreads();

    // ---- load q B-fragment (N = lane%16; lanes>=16 read zero pad e=16..31) ----
    const int qb = (wave * 16 + nloc) * 4 + 2 * hi;
    const v16h qf = SHUF16(q_h8[qb], q_h8[qb + 1]);

    // ---- GEMM1: 8 tiles of logits^T (s rows, n cols) ----
    v8f acc[8];
#pragma unroll
    for (int t = 0; t < 8; ++t) {
        int s = t * 16 + nloc;
        int kb = s * 4 + 2 * hi;                   // chunk0: e=16*hi..16*hi+7 ; chunk1: +8
        v16h af = SHUF16(k_h8[kb], k_h8[kb + 1]);
        v8f z = {};
        acc[t] = __builtin_amdgcn_wmma_f32_16x16x32_f16(false, af, false, qf,
                                                        (short)0, z, false, false);
    }

    // ---- softmax over s (128 values per column n, split across lane pair L, L^16) ----
    float m = -1e30f;
#pragma unroll
    for (int t = 0; t < 8; ++t)
#pragma unroll
        for (int r = 0; r < 8; ++r) m = fmaxf(m, acc[t][r]);
    m = fmaxf(m, __shfl_xor(m, 16, 32));
    float ssum = 0.f;
#pragma unroll
    for (int t = 0; t < 8; ++t)
#pragma unroll
        for (int r = 0; r < 8; ++r) {
            float e = __expf(acc[t][r] - m);
            acc[t][r] = e; ssum += e;
        }
    ssum += __shfl_xor(ssum, 16, 32);
    const float inv = 1.0f / ssum;

    // ---- GEMM2: out^T = V * P, accumulate over 4 K-blocks of 32 ----
    v8f oacc = {};
#pragma unroll
    for (int u = 0; u < 4; ++u) {
        // P from C-layout -> B-layout: one partner exchange (registers only)
        v16h pf;
#pragma unroll
        for (int r = 0; r < 8; ++r) {
            float send = hi ? acc[2 * u][r] : acc[2 * u + 1][r];
            float recv = __shfl_xor(send, 16, 32);
            float own  = hi ? acc[2 * u + 1][r] : acc[2 * u][r];
            pf[r]     = (_Float16)(hi ? recv : own);
            pf[8 + r] = (_Float16)(hi ? own  : recv);
        }
        // V A-fragment: two b128 chunks from v_h[d][s]
        int vb = nloc * 16 + 4 * u + 2 * hi;
        v16h vf = SHUF16(v_h8[vb], v_h8[vb + 1]);
        oacc = __builtin_amdgcn_wmma_f32_16x16x32_f16(false, vf, false, pf,
                                                      (short)0, oacc, false, false);
    }
#pragma unroll
    for (int r = 0; r < 8; ++r) oacc[r] *= inv;

    // ---- GEMM3: out32 = out * w_out^T. A comes straight from oacc (register selects) ----
    v16h of;
#pragma unroll
    for (int j = 0; j < 8; ++j) {
        int kb = ((j < 4) ? 2 * j : 16 + 2 * (j - 4)) + 8 * hi;
        int base = 8 * hi;
        float a0 = (kb     < DV) ? oacc[kb - base]     : 0.f;
        float a1 = (kb + 1 < DV) ? oacc[kb + 1 - base] : 0.f;
        of[2 * j]     = (_Float16)a0;
        of[2 * j + 1] = (_Float16)a1;
    }
#pragma unroll
    for (int tc = 0; tc < 2; ++tc) {
        int c = tc * 16 + nloc;
        int wb = c * 4 + hi;
        v16h wf = SHUF16(wo_h8[wb], wo_h8[wb + 2]);
        v8f z = {};
        v8f o32 = __builtin_amdgcn_wmma_f32_16x16x32_f16(false, of, false, wf,
                                                         (short)0, z, false, false);
        // D layout: lane holds column c, rows n = r + 8*hi. [b][n][c] -> coalesced 16-lane stores
        int nb = n0 + 8 * hi;
#pragma unroll
        for (int r = 0; r < 8; ++r) {
            pout[((size_t)b * NQ + nb + r) * CCV + c] = o32[r];
        }
    }
}

// ------------------------------------------------------------------
// y = cost + gamma * bf16(upsample(out32)); also reduce sum/sumsq of y per batch
__global__ __launch_bounds__(256) void k_compose(const float* __restrict__ cost,
                                                 const float* __restrict__ pout,
                                                 const float* __restrict__ gamma_p,
                                                 float* __restrict__ y,
                                                 float* __restrict__ stats) {
    int b = blockIdx.y;
    float gamma = *gamma_p;
    float s = 0.f, s2 = 0.f;
    for (size_t i = (size_t)blockIdx.x * blockDim.x + threadIdx.x; i < PER_BATCH;
         i += (size_t)gridDim.x * blockDim.x) {
        int c   = (int)(i / (DD * HH * WW));
        int rem = (int)(i % (DD * HH * WW));
        int d = rem / (HH * WW); int rem2 = rem % (HH * WW);
        int h = rem2 / WW;       int w = rem2 % WW;
        // half-pixel bilinear (depth is identity)
        float chf = 0.5f * h - 0.25f;
        float cwf = 0.5f * w - 0.25f;
        int h0 = (int)floorf(chf); float fh = chf - (float)h0;
        int w0 = (int)floorf(cwf); float fw = cwf - (float)w0;
        int h0c = min(max(h0, 0), HQ - 1), h1c = min(max(h0 + 1, 0), HQ - 1);
        int w0c = min(max(w0, 0), WQ - 1), w1c = min(max(w0 + 1, 0), WQ - 1);
        const float* pb = pout + ((size_t)b * NQ + (size_t)d * HQ * WQ) * CCV + c;
        float v00 = pb[(size_t)(h0c * WQ + w0c) * CCV];
        float v01 = pb[(size_t)(h0c * WQ + w1c) * CCV];
        float v10 = pb[(size_t)(h1c * WQ + w0c) * CCV];
        float v11 = pb[(size_t)(h1c * WQ + w1c) * CCV];
        float up = (1.f - fh) * ((1.f - fw) * v00 + fw * v01)
                 +         fh * ((1.f - fw) * v10 + fw * v11);
        up = __bfloat162float(__float2bfloat16(up));   // replicate bf16 cast
        float yv = cost[(size_t)b * PER_BATCH + i] + gamma * up;
        y[(size_t)b * PER_BATCH + i] = yv;
        s += yv; s2 += yv * yv;
    }
    __shared__ float sh[256], sh2[256];
    sh[threadIdx.x] = s; sh2[threadIdx.x] = s2; __syncthreads();
    for (int o = 128; o > 0; o >>= 1) {
        if (threadIdx.x < (unsigned)o) { sh[threadIdx.x] += sh[threadIdx.x + o];
                                         sh2[threadIdx.x] += sh2[threadIdx.x + o]; }
        __syncthreads();
    }
    if (threadIdx.x == 0) {
        atomicAdd(&stats[2 * b],     sh[0]);
        atomicAdd(&stats[2 * b + 1], sh2[0]);
    }
}

// final GroupNorm over y, in place on d_out
__global__ __launch_bounds__(256) void k_norm_out(float* __restrict__ y,
                                                  const float* __restrict__ stats,
                                                  const float* __restrict__ gw,
                                                  const float* __restrict__ gb) {
    const size_t total = (size_t)B_ * PER_BATCH;
    for (size_t idx = (size_t)blockIdx.x * blockDim.x + threadIdx.x; idx < total;
         idx += (size_t)gridDim.x * blockDim.x) {
        int b = (int)(idx / PER_BATCH);
        int c = (int)((idx / (DD * HH * WW)) % CCV);
        float cntf = (float)PER_BATCH;
        float mean = stats[2 * b] / cntf;
        float var  = stats[2 * b + 1] / cntf - mean * mean;
        float rs   = rsqrtf(var + EPSF);
        y[idx] = (y[idx] - mean) * rs * gw[c] + gb[c];
    }
}

// ------------------------------------------------------------------
extern "C" void kernel_launch(void* const* d_in, const int* in_sizes, int n_in,
                              void* d_out, int out_size, void* d_ws, size_t ws_size,
                              hipStream_t stream) {
    const float* cost   = (const float*)d_in[0];
    const float* feat   = (const float*)d_in[1];
    const float* w_q    = (const float*)d_in[2];
    const float* w_k    = (const float*)d_in[3];
    const float* w_v    = (const float*)d_in[4];
    const float* w_out  = (const float*)d_in[5];
    const float* gn_iw  = (const float*)d_in[6];
    const float* gn_ib  = (const float*)d_in[7];
    const float* gn_ow  = (const float*)d_in[8];
    const float* gn_ob  = (const float*)d_in[9];
    const float* gamma  = (const float*)d_in[10];

    float* ws       = (float*)d_ws;
    float* stats_in = ws + WS_STATS_IN;
    float* stats_out= ws + WS_STATS_OUT;
    float* kbuf     = ws + WS_K;
    float* vbuf     = ws + WS_V;
    float* pout     = ws + WS_POUT;
    float* y        = (float*)d_out;

    k_init<<<1, 32, 0, stream>>>(ws);
    k_reduce_in<<<dim3(256, B_), 256, 0, stream>>>(cost, stats_in);
    k_kv<<<(2 * B_ * DQK * NS + 255) / 256, 256, 0, stream>>>(feat, w_k, w_v, kbuf, vbuf);
    // 4 waves/block, 16 query rows per wave: B*NQ/16 / 4 blocks
    k_attn<<<(B_ * (NQ / 16)) / 4, 128, 0, stream>>>(cost, w_q, w_out, gn_iw, gn_ib,
                                                     stats_in, kbuf, vbuf, pout);
    k_compose<<<dim3(512, B_), 256, 0, stream>>>(cost, pout, gamma, y, stats_out);
    k_norm_out<<<2048, 256, 0, stream>>>(y, stats_out, gn_ow, gn_ob);
}
